// TacotronAttention_68771016344043
// MI455X (gfx1250) — matrix operand order repaired and checked by reading, not compiled, not run on hardware
//
#include <hip/hip_runtime.h>
#include <cstdint>
#include <cstddef>

// Problem sizes
#define BB   64
#define TT   2048
#define CC   512
#define AA   192
#define RNN  1024
#define CHUNKS 4            // workgroups per batch
#define TCH  (TT/CHUNKS)    // 512 T-rows per workgroup
#define ROWS 64             // T-rows per LDS tile (4 row-groups x 16)
#define ITERS (TCH/ROWS)    // 8 tiles per workgroup
#define LSTR 516            // padded LDS row stride in floats (bank-conflict-free)
#define NF   (AA/16)        // 12 N-tiles total
#define NH   6              // N-tiles per wave (N split across 2 wave groups)
#define KSN  (CC/32)        // 16 K-steps of 32
#define BUFF (ROWS*LSTR)    // floats per LDS tile buffer (132096 B)

typedef __attribute__((ext_vector_type(8)))  float        v8f;
typedef __attribute__((ext_vector_type(8)))  __bf16       v8bf;
typedef __attribute__((ext_vector_type(16))) __bf16       v16bf;
typedef __attribute__((ext_vector_type(8)))  unsigned int v8u;

// Native CDNA5 V_TANH_F32 if the toolchain exposes it; OCML fallback otherwise.
#if __has_builtin(__builtin_amdgcn_tanhf)
#define FAST_TANH(x) __builtin_amdgcn_tanhf(x)
#else
#define FAST_TANH(x) tanhf(x)
#endif

// Workspace byte offsets (total ~1.24 MB)
#define WS_Q    ((size_t)0)                    // q = enc_word@Wq : 64*192 f32
#define WS_WMB  ((size_t)49152)                // packed bf16 Wm frags: 192*32*16 u16
#define WS_E    ((size_t)(49152+196608))       // raw energies: 64*2048 f32
#define WS_ACC  ((size_t)(WS_E+524288))        // partial weighted ctx: 64*4*512 f32
#define WS_ML   ((size_t)(WS_ACC+524288))      // per-chunk (m,l): 64*4*2 f32

// Build one 16x32 bf16 A fragment (this lane's slice) from an LDS f32 row,
// using the hardware f32->bf16 converter (v_cvt_pk_bf16_f32).
static __device__ __forceinline__ v16bf packA(const float* ap, int base) {
    float4 a0 = *(const float4*)(ap + base);
    float4 a1 = *(const float4*)(ap + base + 4);
    float4 b0 = *(const float4*)(ap + base + 8);
    float4 b1 = *(const float4*)(ap + base + 12);
    v8f lo = {a0.x, a0.y, a0.z, a0.w, a1.x, a1.y, a1.z, a1.w};
    v8f hi = {b0.x, b0.y, b0.z, b0.w, b1.x, b1.y, b1.z, b1.w};
    v8bf l = __builtin_convertvector(lo, v8bf);
    v8bf h = __builtin_convertvector(hi, v8bf);
    return __builtin_shufflevector(l, h, 0, 1, 2, 3, 4, 5, 6, 7,
                                   8, 9, 10, 11, 12, 13, 14, 15);
}

// ---------------- Prep kernel 1: q[b][a] = enc_word[b,:] @ Wq[:,a] ----------------
__global__ void tac_prep_q(const float* __restrict__ enc, const float* __restrict__ Wq,
                           float* __restrict__ q) {
    int idx = blockIdx.x * blockDim.x + threadIdx.x;
    if (idx >= BB * AA) return;
    int b = idx / AA, a = idx % AA;
    const float* e = enc + (size_t)b * RNN;
    float s = 0.f;
    for (int k = 0; k < RNN; ++k) s += e[k] * Wq[(size_t)k * AA + a];
    q[idx] = s;
}

// ---- Prep kernel 2: pack Wm into WMMA B-fragment layout (bf16, 32 B per lane) ----
// frag = n*16 + ks covers K=[ks*32, +32), N=[n*16, +16)
// lane l: column N = n*16 + (l&15); K = ks*32 + (l<16?0:16) + j, j=0..15
__global__ void tac_prep_wmb(const float* __restrict__ Wm, unsigned short* __restrict__ out) {
    int idx = blockIdx.x * blockDim.x + threadIdx.x;
    if (idx >= NF * KSN * 32) return;
    int frag = idx >> 5, lane = idx & 31;
    int n = frag / KSN, ks = frag % KSN;
    int col = n * 16 + (lane & 15);
    int k0 = ks * 32 + ((lane < 16) ? 0 : 16);
    unsigned short* d = out + (size_t)idx * 16;
    for (int j = 0; j < 16; ++j) {
        __bf16 h = (__bf16)Wm[(size_t)(k0 + j) * AA + col];
        d[j] = __builtin_bit_cast(unsigned short, h);
    }
}

// ---------------- Main fused kernel: one WG = one (batch, T-chunk) ----------------
// Double-buffered 64x512 f32 context tiles in LDS, staged by async global->LDS DMA
// overlapped with bf16 WMMA + online softmax on the other buffer.
__global__ void tac_attn_main(const float* __restrict__ gctx,
                              const float* __restrict__ qbuf,
                              const float* __restrict__ vvec,
                              const unsigned short* __restrict__ wmbp,
                              float* __restrict__ ws_e,
                              float* __restrict__ ws_acc,
                              float* __restrict__ ws_ml) {
    extern __shared__ float smem[];
    // LDS layout (byte offsets match async-LDS addressing below):
    //   [0, 132096)        buf0: 64 x 516 f32
    //   [132096, 264192)   buf1: 64 x 516 f32
    //   [264192, 264704)   epart[2][64] f32
    float* epart = smem + 2 * BUFF;

    const int tid  = threadIdx.x;
    const int lane = tid & 31;
    const int wv   = tid >> 5;                 // 8 waves
    const int rgrp = wv & 3;                   // row group: rows [16*rgrp, +16)
    const int nh   = wv >> 2;                  // N-half: n-tiles [6*nh, +6)
    const int nh6  = nh * NH;
    const int b     = blockIdx.x >> 2;
    const int chunk = blockIdx.x & 3;
    const int tchunk0 = chunk * TCH;

    // per-lane q + v for the energy dot (column a = 16*(nh6+n) + lane%16)
    float ql[NH], vl[NH];
#pragma unroll
    for (int n = 0; n < NH; ++n) {
        int a = (nh6 + n) * 16 + (lane & 15);
        ql[n] = qbuf[b * AA + a];
        vl[n] = vvec[a];
    }

    const v8u* wbl = (const v8u*)wmbp + (size_t)nh6 * KSN * 32 + lane;
    const int koff = (lane < 16) ? 0 : 16;
    const int arow = rgrp * 16 + (lane & 15);

    float m_run = -3.0e38f, l_run = 0.f;
    float acc0 = 0.f, acc1 = 0.f;              // running weighted ctx, cols 2*tid(+1)
    const int c0 = 2 * tid;

    // ---- async-stage one 64x512 f32 tile into LDS buffer `whichbuf` ----
    auto stage = [&](int whichbuf, int iter) {
        const float* gsrc = gctx + ((size_t)b * TT + tchunk0 + iter * ROWS) * CC;
        const unsigned bufbase = (unsigned)(whichbuf * BUFF * 4);
        for (int j = 0; j < 32; ++j) {         // 8192 float4 slots / 256 threads
            int s   = j * 256 + tid;
            int row = s >> 7;
            int c4  = (s & 127) << 2;
            unsigned lds_b = bufbase + (unsigned)((row * LSTR + c4) * 4);
            unsigned long long ga =
                (unsigned long long)(uintptr_t)(gsrc + (size_t)row * CC + c4);
            asm volatile("global_load_async_to_lds_b128 %0, %1, off"
                         :: "v"(lds_b), "v"(ga) : "memory");
        }
    };

    stage(0, 0);
    asm volatile("s_wait_asynccnt 0" ::: "memory");
    __syncthreads();

    for (int iter = 0; iter < ITERS; ++iter) {
        const float* cb = smem + (iter & 1) * BUFF;   // compute buffer
        if (iter + 1 < ITERS) stage((iter + 1) & 1, iter + 1);  // overlap DMA

        // ---- bf16 WMMA: pc[16 rows, 96 cols] for this wave's (rows, N-half) ----
        v8f acc[NH];
        const v8f zz = {0.f,0.f,0.f,0.f,0.f,0.f,0.f,0.f};
#pragma unroll
        for (int n = 0; n < NH; ++n) acc[n] = zz;

        const float* ap = cb + arow * LSTR;
        for (int ks = 0; ks < KSN; ++ks) {
            v8u bf[NH];
#pragma unroll
            for (int n = 0; n < NH; ++n)       // one 12-load clause
                bf[n] = wbl[(n * KSN + ks) * 32];
            v16bf va = packA(ap, ks * 32 + koff);   // ds loads + cvts hide latency
#pragma unroll
            for (int n = 0; n < NH; ++n)
                acc[n] = __builtin_amdgcn_wmma_f32_16x16x32_bf16(
                    false, va, false, __builtin_bit_cast(v16bf, bf[n]),
                    (short)0, acc[n], false, false);
        }

        // ---- partial energies: tanh(q+pc)*v, butterfly reduce over 16 lanes ----
#pragma unroll
        for (int r = 0; r < 8; ++r) {
            float e = 0.f;
#pragma unroll
            for (int n = 0; n < NH; ++n) e += FAST_TANH(acc[n][r] + ql[n]) * vl[n];
            e += __shfl_xor(e, 1); e += __shfl_xor(e, 2);
            e += __shfl_xor(e, 4); e += __shfl_xor(e, 8);
            int slot = rgrp * 16 + ((lane < 16) ? r : r + 8);
            if ((lane & 15) == 0) epart[nh * ROWS + slot] = e;
        }
        __syncthreads();

        // ---- online softmax + weighted-ctx accumulate (redundant scalar state) ----
        const float* ep0 = epart;
        const float* ep1 = epart + ROWS;
        if (tid < ROWS)
            ws_e[(size_t)b * TT + tchunk0 + iter * ROWS + tid] = ep0[tid] + ep1[tid];
        float mt = -3.0e38f;
        for (int j = 0; j < ROWS; ++j) mt = fmaxf(mt, ep0[j] + ep1[j]);
        float mnew = fmaxf(m_run, mt);
        float sc = __expf(m_run - mnew);
        acc0 *= sc; acc1 *= sc; l_run *= sc;
        for (int j = 0; j < ROWS; ++j) {
            float p = __expf(ep0[j] + ep1[j] - mnew);   // args <= 0: fast exp is safe
            l_run += p;
            float2 cv = *(const float2*)(cb + j * LSTR + c0);
            acc0 += p * cv.x; acc1 += p * cv.y;
        }
        m_run = mnew;

        asm volatile("s_wait_asynccnt 0" ::: "memory");  // next tile DMA done
        __syncthreads();
    }

    // ---- write chunk partials ----
    float* pa = ws_acc + ((size_t)(b * CHUNKS + chunk)) * CC;
    pa[c0] = acc0; pa[c0 + 1] = acc1;
    if (tid == 0) {
        ws_ml[(b * CHUNKS + chunk) * 2 + 0] = m_run;
        ws_ml[(b * CHUNKS + chunk) * 2 + 1] = l_run;
    }
}

// -------- Combine: merge 4 softmax partials per batch, emit ctx + weights --------
__global__ void tac_combine(const float* __restrict__ ws_ml,
                            const float* __restrict__ ws_acc,
                            const float* __restrict__ ws_e,
                            float* __restrict__ out) {
    int b = blockIdx.x, tid = threadIdx.x;
    float m[CHUNKS], l[CHUNKS];
    float ms = -3.0e38f;
    for (int i = 0; i < CHUNKS; ++i) {
        m[i] = ws_ml[(b * CHUNKS + i) * 2 + 0];
        l[i] = ws_ml[(b * CHUNKS + i) * 2 + 1];
        ms = fmaxf(ms, m[i]);
    }
    float L = 0.f, si[CHUNKS];
    for (int i = 0; i < CHUNKS; ++i) { si[i] = __expf(m[i] - ms); L += l[i] * si[i]; }
    float inv = 1.f / L;
    // attention_context: out[0 .. B*C)
    for (int c = tid; c < CC; c += 256) {
        float s = 0.f;
        for (int i = 0; i < CHUNKS; ++i)
            s += ws_acc[((size_t)(b * CHUNKS + i)) * CC + c] * si[i];
        out[(size_t)b * CC + c] = s * inv;
    }
    // attention_weights: out[B*C .. B*C + B*T)
    const float* eb = ws_e + (size_t)b * TT;
    float* wout = out + (size_t)BB * CC + (size_t)b * TT;
    for (int t = tid; t < TT; t += 256) wout[t] = __expf(eb[t] - ms) * inv;
}

extern "C" void kernel_launch(void* const* d_in, const int* in_sizes, int n_in,
                              void* d_out, int out_size, void* d_ws, size_t ws_size,
                              hipStream_t stream) {
    (void)in_sizes; (void)n_in; (void)out_size; (void)ws_size;
    const float* enc = (const float*)d_in[0];
    const float* ctx = (const float*)d_in[1];
    const float* Wq  = (const float*)d_in[2];
    const float* Wm  = (const float*)d_in[3];
    const float* v   = (const float*)d_in[4];
    float* out = (float*)d_out;

    char* ws = (char*)d_ws;
    float*          ws_q   = (float*)(ws + WS_Q);
    unsigned short* ws_wmb = (unsigned short*)(ws + WS_WMB);
    float*          ws_e   = (float*)(ws + WS_E);
    float*          ws_acc = (float*)(ws + WS_ACC);
    float*          ws_ml  = (float*)(ws + WS_ML);

    tac_prep_q<<<(BB * AA + 255) / 256, 256, 0, stream>>>(enc, Wq, ws_q);
    tac_prep_wmb<<<(NF * KSN * 32 + 255) / 256, 256, 0, stream>>>(Wm, ws_wmb);

    size_t shmem = (size_t)(2 * BUFF + 2 * ROWS) * sizeof(float);   // 264704 B
    tac_attn_main<<<BB * CHUNKS, 256, shmem, stream>>>(ctx, ws_q, v, ws_wmb,
                                                       ws_e, ws_acc, ws_ml);
    tac_combine<<<BB, 256, 0, stream>>>(ws_ml, ws_acc, ws_e, out);
}
